// PokerMambaClassifier_26568667693072
// MI455X (gfx1250) — compile-verified, matrix operand order checked by reading, not run on hardware
//
#include <hip/hip_runtime.h>
#include <hip/hip_bf16.h>
#include <stdint.h>

#define D_MODEL   512
#define N_LAYERS  4
#define D_INNER   1024
#define D_STATE   16
#define D_CONV    4
#define DT_RANK   32
#define N_CAT     4
#define CARD      52
#define EMBED_DIM 32
#define N_FEAT    64
#define N_CLASSES 10
#define BB        4
#define TT        1024
#define NROW      (BB*TT)               /* 4096 */
#define DBC_W     (DT_RANK + 2*D_STATE) /* 64 */

typedef __attribute__((ext_vector_type(16))) __bf16 v16bf;
typedef __attribute__((ext_vector_type(8)))  __bf16 v8bf;
typedef __attribute__((ext_vector_type(8)))  float  v8f;
typedef int vint4 __attribute__((__vector_size__(4 * sizeof(int))));

__device__ __forceinline__ float siluf(float x) { return x / (1.f + __expf(-x)); }

// ---- async global->LDS copy (CDNA5 GLOBAL_LOAD_ASYNC_TO_LDS_B128), guarded ----
#if defined(__has_builtin)
#if __has_builtin(__builtin_amdgcn_global_load_async_to_lds_b128) && \
    __has_builtin(__builtin_amdgcn_s_wait_asynccnt)
#define HAVE_ASYNC_LDS 1
#endif
#endif

__device__ __forceinline__ void lds_copy16(const __bf16* g, __bf16* l) {
#ifdef HAVE_ASYNC_LDS
    __builtin_amdgcn_global_load_async_to_lds_b128(
        (__attribute__((address_space(1))) vint4*)
            (__attribute__((address_space(1))) void*)(void*)const_cast<__bf16*>(g),
        (__attribute__((address_space(3))) vint4*)
            (__attribute__((address_space(3))) void*)(void*)l,
        0, 0);
#else
    *(v8bf*)l = *(const v8bf*)g;
#endif
}

__device__ __forceinline__ void lds_copy_wait() {
#ifdef HAVE_ASYNC_LDS
    __builtin_amdgcn_s_wait_asynccnt(0);
#endif
}

// ---------------------------------------------------------------- convert
__global__ void f32_to_bf16_kernel(const float* __restrict__ in,
                                   __bf16* __restrict__ out, int n) {
    int i = blockIdx.x * blockDim.x + threadIdx.x;
    if (i < n) out[i] = (__bf16)in[i];
}

// ---------------------------------------------------------------- embedding + cont proj
__global__ void embed_kernel(const float* __restrict__ X,
                             const float* __restrict__ emb,
                             const float* __restrict__ cw,
                             const float* __restrict__ cb,
                             float* __restrict__ H) {
    int idx = blockIdx.x * blockDim.x + threadIdx.x;
    if (idx >= NROW * D_MODEL) return;
    int row = idx / D_MODEL, col = idx % D_MODEL;
    const float* xr = X + row * N_FEAT;
    float v;
    if (col < N_CAT * EMBED_DIM) {
        int j = col / EMBED_DIM, e = col % EMBED_DIM;
        int id = (int)xr[j];
        id = id < 0 ? 0 : (id >= CARD ? CARD - 1 : id);
        v = emb[(j * CARD + id) * EMBED_DIM + e];
    } else {
        int c = col - N_CAT * EMBED_DIM;
        const float* wr = cw + c * (N_FEAT - N_CAT);
        float acc = cb[c];
        #pragma unroll 4
        for (int k = 0; k < N_FEAT - N_CAT; ++k) acc += xr[N_CAT + k] * wr[k];
        v = acc;
    }
    H[idx] = v;
}

// ---------------------------------------------------------------- RMSNorm (wave32 per row)
__global__ void rmsnorm_kernel(const float* __restrict__ H,
                               const float* __restrict__ w,
                               __bf16* __restrict__ out) {
    int wave = threadIdx.x >> 5, lane = threadIdx.x & 31;
    int row = blockIdx.x * 8 + wave;
    const float* hr = H + row * D_MODEL;
    float vals[16];
    float ss = 0.f;
    #pragma unroll
    for (int i = 0; i < 16; ++i) { float v = hr[lane + i * 32]; vals[i] = v; ss += v * v; }
    #pragma unroll
    for (int off = 16; off >= 1; off >>= 1) ss += __shfl_xor(ss, off, 32);
    float inv = rsqrtf(ss / (float)D_MODEL + 1e-5f);
    __bf16* orow = out + row * D_MODEL;
    #pragma unroll
    for (int i = 0; i < 16; ++i) { int c = lane + i * 32; orow[c] = (__bf16)(vals[i] * inv * w[c]); }
}

// ---------------------------------------------------------------- WMMA GEMM
// C[M,N] (+)= A[M,K] @ W[N,K]^T , A/W bf16 row-major, C f32.
// M%128==0, N%64==0, K%32==0. Block: 256 thr = 8 waves; block tile 128x64;
// waves arranged 4x2, each wave owns 32x32 (2 A-frags x 2 B-frags -> 4 WMMA,
// 2 ds_load_b128 per v_wmma_f32_16x16x32_bf16).
#define LDT 40
__global__ void gemm_bf16_wmma(const __bf16* __restrict__ A,
                               const __bf16* __restrict__ W,
                               float* __restrict__ C,
                               int M, int N, int K, int addTo) {
    __shared__ __align__(16) __bf16 lA[128 * LDT];
    __shared__ __align__(16) __bf16 lB[64 * LDT];
    const int tid  = threadIdx.x;
    const int wave = tid >> 5, lane = tid & 31;
    const int half = lane >> 4, ln = lane & 15;
    const int wm = wave >> 1, wn = wave & 1;           // 4x2 wave grid
    const int bM = blockIdx.y * 128, bN = blockIdx.x * 64;

    // A tile: 128x32 -> 512 16B-chunks, 2 per thread. B tile: 64x32 -> 256 chunks.
    const int ar0 = (tid * 2) >> 2,  ac0 = ((tid * 2) & 3) * 8;
    const int ar1 = (tid * 2 + 1) >> 2, ac1 = ((tid * 2 + 1) & 3) * 8;
    const int br  = tid >> 2, bc = (tid & 3) * 8;

    v8f acc00 = {}; v8f acc01 = {}; v8f acc10 = {}; v8f acc11 = {};

    for (int kt = 0; kt < K; kt += 32) {
        __syncthreads();
        lds_copy16(&A[(size_t)(bM + ar0) * K + kt + ac0], &lA[ar0 * LDT + ac0]);
        lds_copy16(&A[(size_t)(bM + ar1) * K + kt + ac1], &lA[ar1 * LDT + ac1]);
        lds_copy16(&W[(size_t)(bN + br)  * K + kt + bc],  &lB[br  * LDT + bc]);
        if (kt + 32 < K) {
            __builtin_prefetch(&A[(size_t)(bM + ar0) * K + kt + 32 + ac0], 0, 0);
            __builtin_prefetch(&W[(size_t)(bN + br)  * K + kt + 32 + bc],  0, 0);
        }
        lds_copy_wait();
        __syncthreads();

        // B fragments: rows = columns n of output (lane ln), half selects K half
        v16bf bf0, bf1;
        {
            v8bf x0 = *(const v8bf*)&lB[(wn * 32 + ln) * LDT + half * 8];
            v8bf x1 = *(const v8bf*)&lB[(wn * 32 + ln) * LDT + 16 + half * 8];
            v8bf y0 = *(const v8bf*)&lB[(wn * 32 + 16 + ln) * LDT + half * 8];
            v8bf y1 = *(const v8bf*)&lB[(wn * 32 + 16 + ln) * LDT + 16 + half * 8];
            #pragma unroll
            for (int j = 0; j < 8; ++j) {
                bf0[j] = x0[j]; bf0[8 + j] = x1[j];
                bf1[j] = y0[j]; bf1[8 + j] = y1[j];
            }
        }
        // A fragments
        v16bf af0, af1;
        {
            v8bf x0 = *(const v8bf*)&lA[(wm * 32 + ln) * LDT + half * 8];
            v8bf x1 = *(const v8bf*)&lA[(wm * 32 + ln) * LDT + 16 + half * 8];
            v8bf y0 = *(const v8bf*)&lA[(wm * 32 + 16 + ln) * LDT + half * 8];
            v8bf y1 = *(const v8bf*)&lA[(wm * 32 + 16 + ln) * LDT + 16 + half * 8];
            #pragma unroll
            for (int j = 0; j < 8; ++j) {
                af0[j] = x0[j]; af0[8 + j] = x1[j];
                af1[j] = y0[j]; af1[8 + j] = y1[j];
            }
        }

        acc00 = __builtin_amdgcn_wmma_f32_16x16x32_bf16(false, af0, false, bf0, (short)0, acc00, false, false);
        acc01 = __builtin_amdgcn_wmma_f32_16x16x32_bf16(false, af0, false, bf1, (short)0, acc01, false, false);
        acc10 = __builtin_amdgcn_wmma_f32_16x16x32_bf16(false, af1, false, bf0, (short)0, acc10, false, false);
        acc11 = __builtin_amdgcn_wmma_f32_16x16x32_bf16(false, af1, false, bf1, (short)0, acc11, false, false);
    }

    // Store: C/D layout — VGPR r holds (M = r + 8*half, N = ln)
    #pragma unroll
    for (int sm = 0; sm < 2; ++sm) {
        #pragma unroll
        for (int sn = 0; sn < 2; ++sn) {
            v8f acc = sm ? (sn ? acc11 : acc10) : (sn ? acc01 : acc00);
            int gr0 = bM + wm * 32 + sm * 16 + half * 8;
            int gc  = bN + wn * 32 + sn * 16 + ln;
            #pragma unroll
            for (int r = 0; r < 8; ++r) {
                float* p = &C[(size_t)(gr0 + r) * N + gc];
                if (addTo) *p += acc[r]; else *p = acc[r];
            }
        }
    }
}

// ---------------------------------------------------------------- causal depthwise conv + SiLU
__global__ void conv_silu_kernel(const float* __restrict__ XZ,
                                 const float* __restrict__ cw,
                                 const float* __restrict__ cb,
                                 float* __restrict__ XC,
                                 __bf16* __restrict__ XC16) {
    int idx = blockIdx.x * blockDim.x + threadIdx.x;
    if (idx >= NROW * D_INNER) return;
    int row = idx / D_INNER, d = idx % D_INNER;
    int b = row / TT, t = row % TT;
    const float* w4 = cw + d * D_CONV;
    float acc = cb[d];
    #pragma unroll
    for (int j = 0; j < D_CONV; ++j) {
        int tt = t - (D_CONV - 1) + j;
        float xv = (tt >= 0) ? XZ[(size_t)(b * TT + tt) * (2 * D_INNER) + d] : 0.f;
        acc += w4[j] * xv;
    }
    float y = siluf(acc);
    XC[idx] = y;
    XC16[idx] = (__bf16)y;
}

// ---------------------------------------------------------------- dbc[:, :DT_RANK] -> bf16
__global__ void dbc_split_kernel(const float* __restrict__ DBC, __bf16* __restrict__ A16) {
    int idx = blockIdx.x * blockDim.x + threadIdx.x;
    if (idx >= NROW * DT_RANK) return;
    int row = idx / DT_RANK, k = idx % DT_RANK;
    A16[idx] = (__bf16)DBC[(size_t)row * DBC_W + k];
}

// ---------------------------------------------------------------- selective scan (sequential in T)
__global__ void scan_kernel(const float* __restrict__ DELTA,
                            const float* __restrict__ XC,
                            const float* __restrict__ DBC,
                            const float* __restrict__ A_log,
                            const float* __restrict__ dtb,
                            float* __restrict__ Y) {
    int b = blockIdx.x >> 2;
    int d = (blockIdx.x & 3) * 256 + threadIdx.x;
    float Acoef[D_STATE];
    #pragma unroll
    for (int n = 0; n < D_STATE; ++n) Acoef[n] = -__expf(A_log[d * D_STATE + n]);
    float bias = dtb[d];
    float h[D_STATE];
    #pragma unroll
    for (int n = 0; n < D_STATE; ++n) h[n] = 0.f;

    __shared__ float sBC[2 * D_STATE];
    for (int t = 0; t < TT; ++t) {
        int row = b * TT + t;
        if (threadIdx.x < 2 * D_STATE)
            sBC[threadIdx.x] = DBC[(size_t)row * DBC_W + DT_RANK + threadIdx.x];
        __syncthreads();
        float dl = DELTA[(size_t)row * D_INNER + d] + bias;
        float delta = (dl > 20.f) ? dl : log1pf(__expf(dl));
        float du = delta * XC[(size_t)row * D_INNER + d];
        float y = 0.f;
        #pragma unroll
        for (int n = 0; n < D_STATE; ++n) {
            h[n] = __expf(delta * Acoef[n]) * h[n] + du * sBC[n];
            y += h[n] * sBC[D_STATE + n];
        }
        Y[(size_t)row * D_INNER + d] = y;
        __syncthreads();
    }
}

// ---------------------------------------------------------------- y = (y + D*xc) * silu(z) -> bf16
__global__ void ypost_kernel(const float* __restrict__ Y,
                             const float* __restrict__ XC,
                             const float* __restrict__ XZ,
                             const float* __restrict__ Dskip,
                             __bf16* __restrict__ Y16) {
    int idx = blockIdx.x * blockDim.x + threadIdx.x;
    if (idx >= NROW * D_INNER) return;
    int row = idx / D_INNER, d = idx % D_INNER;
    float z = XZ[(size_t)row * (2 * D_INNER) + D_INNER + d];
    float y = (Y[idx] + Dskip[d] * XC[idx]) * siluf(z);
    Y16[idx] = (__bf16)y;
}

// ---------------------------------------------------------------- head (N=10, scalar)
__global__ void head_kernel(const float* __restrict__ H,
                            const float* __restrict__ hw,
                            const float* __restrict__ hb,
                            float* __restrict__ out) {
    int idx = blockIdx.x * blockDim.x + threadIdx.x;
    if (idx >= NROW * N_CLASSES) return;
    int row = idx / N_CLASSES, c = idx % N_CLASSES;
    const float* hr = H + (size_t)row * D_MODEL;
    const float* wr = hw + (size_t)c * D_MODEL;
    float acc = hb[c];
    #pragma unroll 4
    for (int k = 0; k < D_MODEL; ++k) acc += hr[k] * wr[k];
    out[idx] = acc;
}

// ================================================================ launcher
extern "C" void kernel_launch(void* const* d_in, const int* in_sizes, int n_in,
                              void* d_out, int out_size, void* d_ws, size_t ws_size,
                              hipStream_t stream) {
    (void)in_sizes; (void)n_in; (void)out_size; (void)ws_size;

    const float* X        = (const float*)d_in[0];
    const float* emb      = (const float*)d_in[1];
    const float* cont_w   = (const float*)d_in[2];
    const float* cont_b   = (const float*)d_in[3];
    const float* norm_w   = (const float*)d_in[4];
    const float* in_proj  = (const float*)d_in[5];
    const float* conv_w   = (const float*)d_in[6];
    const float* conv_b   = (const float*)d_in[7];
    const float* x_proj   = (const float*)d_in[8];
    const float* dt_w     = (const float*)d_in[9];
    const float* dt_b     = (const float*)d_in[10];
    const float* A_log    = (const float*)d_in[11];
    const float* D_skip   = (const float*)d_in[12];
    const float* out_proj = (const float*)d_in[13];
    const float* head_w   = (const float*)d_in[14];
    const float* head_b   = (const float*)d_in[15];
    float* out = (float*)d_out;

    char* ws = (char*)d_ws;
    size_t off = 0;
    auto alloc = [&](size_t bytes) -> void* {
        void* p = ws + off;
        off = (off + bytes + 255) & ~(size_t)255;
        return p;
    };

    float*  H     = (float*) alloc((size_t)NROW * D_MODEL * 4);
    float*  XZ    = (float*) alloc((size_t)NROW * 2 * D_INNER * 4);
    float*  XC    = (float*) alloc((size_t)NROW * D_INNER * 4);
    float*  DBC   = (float*) alloc((size_t)NROW * DBC_W * 4);
    float*  DELTA = (float*) alloc((size_t)NROW * D_INNER * 4);
    float*  Y     = (float*) alloc((size_t)NROW * D_INNER * 4);
    __bf16* XN16  = (__bf16*)alloc((size_t)NROW * D_MODEL * 2);
    __bf16* XC16  = (__bf16*)alloc((size_t)NROW * D_INNER * 2);
    __bf16* DA16  = (__bf16*)alloc((size_t)NROW * DT_RANK * 2);
    __bf16* Y16   = (__bf16*)alloc((size_t)NROW * D_INNER * 2);
    __bf16* IPW16 = (__bf16*)alloc((size_t)N_LAYERS * 2 * D_INNER * D_MODEL * 2);
    __bf16* XPW16 = (__bf16*)alloc((size_t)N_LAYERS * DBC_W * D_INNER * 2);
    __bf16* DTW16 = (__bf16*)alloc((size_t)N_LAYERS * D_INNER * DT_RANK * 2);
    __bf16* OPW16 = (__bf16*)alloc((size_t)N_LAYERS * D_MODEL * D_INNER * 2);

    const int TPB = 256;
    auto cvt = [&](const float* src, __bf16* dst, int n) {
        f32_to_bf16_kernel<<<(n + TPB - 1) / TPB, TPB, 0, stream>>>(src, dst, n);
    };
    cvt(in_proj,  IPW16, N_LAYERS * 2 * D_INNER * D_MODEL);
    cvt(x_proj,   XPW16, N_LAYERS * DBC_W * D_INNER);
    cvt(dt_w,     DTW16, N_LAYERS * D_INNER * DT_RANK);
    cvt(out_proj, OPW16, N_LAYERS * D_MODEL * D_INNER);

    embed_kernel<<<(NROW * D_MODEL) / TPB, TPB, 0, stream>>>(X, emb, cont_w, cont_b, H);

    for (int l = 0; l < N_LAYERS; ++l) {
        rmsnorm_kernel<<<NROW / 8, TPB, 0, stream>>>(H, norm_w + l * D_MODEL, XN16);

        // in_proj: [4096,512] x [2048,512]^T -> XZ [4096,2048]
        gemm_bf16_wmma<<<dim3(2 * D_INNER / 64, NROW / 128), TPB, 0, stream>>>(
            XN16, IPW16 + (size_t)l * 2 * D_INNER * D_MODEL, XZ,
            NROW, 2 * D_INNER, D_MODEL, 0);

        conv_silu_kernel<<<(NROW * D_INNER) / TPB, TPB, 0, stream>>>(
            XZ, conv_w + (size_t)l * D_INNER * D_CONV, conv_b + (size_t)l * D_INNER, XC, XC16);

        // x_proj: [4096,1024] x [64,1024]^T -> DBC [4096,64]
        gemm_bf16_wmma<<<dim3(DBC_W / 64, NROW / 128), TPB, 0, stream>>>(
            XC16, XPW16 + (size_t)l * DBC_W * D_INNER, DBC,
            NROW, DBC_W, D_INNER, 0);

        dbc_split_kernel<<<(NROW * DT_RANK) / TPB, TPB, 0, stream>>>(DBC, DA16);

        // dt: [4096,32] x [1024,32]^T -> DELTA [4096,1024]
        gemm_bf16_wmma<<<dim3(D_INNER / 64, NROW / 128), TPB, 0, stream>>>(
            DA16, DTW16 + (size_t)l * D_INNER * DT_RANK, DELTA,
            NROW, D_INNER, DT_RANK, 0);

        scan_kernel<<<BB * (D_INNER / 256), TPB, 0, stream>>>(
            DELTA, XC, DBC, A_log + (size_t)l * D_INNER * D_STATE,
            dt_b + (size_t)l * D_INNER, Y);

        ypost_kernel<<<(NROW * D_INNER) / TPB, TPB, 0, stream>>>(
            Y, XC, XZ, D_skip + (size_t)l * D_INNER, Y16);

        // out_proj (+residual into H): [4096,1024] x [512,1024]^T -> H [4096,512]
        gemm_bf16_wmma<<<dim3(D_MODEL / 64, NROW / 128), TPB, 0, stream>>>(
            Y16, OPW16 + (size_t)l * D_MODEL * D_INNER, H,
            NROW, D_MODEL, D_INNER, 1);
    }

    head_kernel<<<(NROW * N_CLASSES + TPB - 1) / TPB, TPB, 0, stream>>>(H, head_w, head_b, out);
}
